// Swin3DTransformerBlock_82824149336761
// MI455X (gfx1250) — matrix operand order repaired and checked by reading, not compiled, not run on hardware
//
#include <hip/hip_runtime.h>
#include <hip/hip_bf16.h>
#include <hip/hip_fp16.h>

typedef __attribute__((ext_vector_type(16))) _Float16 v16h;
typedef __attribute__((ext_vector_type(8)))  float    v8f;
typedef int vint4 __attribute__((vector_size(16)));

#define B_    2
#define C_    4
#define H_    60
#define W_    120
#define DIM_  512
#define HEADS_ 8
#define DH_   64
#define WS0   2
#define WS1   6
#define WS2   12
#define SS0_  1
#define SS1_  3
#define SS2_  6
#define N_TOK 144
#define C1_   2
#define H1_   10
#define W1_   10
#define NW_   200
#define L_    (C_*H_*W_)       /* 28800 */
#define ROWS  (B_*L_)          /* 57600 */

// ---------------- CDNA5 async global->LDS (ASYNCcnt path) ----------------

#if __has_builtin(__builtin_amdgcn_s_wait_asynccnt)
#define WAIT_ASYNC(n) __builtin_amdgcn_s_wait_asynccnt(n)
#else
#define WAIT_ASYNC(n) asm volatile("s_wait_asynccnt %0" ::"n"(n) : "memory")
#endif

__device__ __forceinline__ void async_copy16(_Float16* dstLds,
                                             const _Float16* srcGlb) {
#if __has_builtin(__builtin_amdgcn_global_load_async_to_lds_b128)
    __builtin_amdgcn_global_load_async_to_lds_b128(
        (__attribute__((address_space(1))) vint4*)(srcGlb),
        (__attribute__((address_space(3))) vint4*)(dstLds), 0, 0);
#else
    *(uint4*)dstLds = *(const uint4*)srcGlb;
#endif
}

// -------------------- helpers --------------------

__device__ __forceinline__ int group_label(int c, int h, int w) {
    // mirrors compute_3d_shifted_window_mask slice labels + longitude-wrap merge
    int ci = (c < C_ - WS0) ? 0 : ((c < C_ - SS0_) ? 1 : 2);
    int hi = (h < H_ - WS1) ? 0 : ((h < H_ - SS1_) ? 1 : 2);
    int wi = (w < W_ - WS2) ? 0 : ((w < W_ - SS2_) ? 1 : 2);
    if (wi == 1) wi = 2;   // warped (periodic longitude) merge
    return ci * 9 + hi * 3 + wi;
}

// A fragment: 16x32 f16 row-major, ldm halves; per-lane = two contiguous 16B runs
__device__ __forceinline__ v16h load_a_frag(const _Float16* base, int ldm, int lane) {
    int m  = lane & 15;
    int hi = lane >> 4;
    const _Float16* r = base + (size_t)m * ldm + hi * 16;
    union { uint4 q[2]; v16h v; } u;
    u.q[0] = *(const uint4*)(r);
    u.q[1] = *(const uint4*)(r + 8);
    return u.v;
}

// B fragment: 32x16 f16 (K rows x N cols) row-major, ldb halves (column-strided)
__device__ __forceinline__ v16h load_b_frag(const _Float16* base, int ldb, int lane) {
    v16h b;
    int n    = lane & 15;
    int koff = (lane >> 4) * 16;
#pragma unroll
    for (int v = 0; v < 8; v++) {
        b[2 * v]     = base[(size_t)(koff + 2 * v) * ldb + n];
        b[2 * v + 1] = base[(size_t)(koff + 2 * v + 1) * ldb + n];
    }
    return b;
}

// -------------------- small kernels --------------------

__global__ void f32_to_f16_kernel(const float* __restrict__ src,
                                  _Float16* __restrict__ dst, int n) {
    int i = blockIdx.x * 256 + threadIdx.x;
    if (i < n) dst[i] = (_Float16)src[i];
}

// LayerNorm; mode 0: gather canvas->window order with cyclic shift; mode 1: identity
__global__ __launch_bounds__(128)
void ln_gather_kernel(const float* __restrict__ x,
                      const float* __restrict__ scale,
                      const float* __restrict__ bias,
                      _Float16* __restrict__ dst, int mode) {
    int row = blockIdx.x;
    int srcRow;
    if (mode == 0) {
        int b    = row / (NW_ * N_TOK);
        int rem  = row % (NW_ * N_TOK);
        int wIdx = rem / N_TOK;
        int p    = rem % N_TOK;
        int c1 = wIdx / (H1_ * W1_);
        int h1 = (wIdx / W1_) % H1_;
        int w1 = wIdx % W1_;
        int i0 = p / (WS1 * WS2);
        int i1 = (p / WS2) % WS1;
        int i2 = p % WS2;
        int cs = c1 * WS0 + i0, hs = h1 * WS1 + i1, ws = w1 * WS2 + i2;
        int c = (cs + SS0_) % C_;
        int h = (hs + SS1_) % H_;
        int w = (ws + SS2_) % W_;
        srcRow = b * L_ + (c * H_ + h) * W_ + w;
    } else {
        srcRow = row;
    }
    const float* xr = x + (size_t)srcRow * DIM_;
    int t = threadIdx.x;
    float4 v4 = *(const float4*)(xr + t * 4);
    float s  = v4.x + v4.y + v4.z + v4.w;
    float sq = v4.x * v4.x + v4.y * v4.y + v4.z * v4.z + v4.w * v4.w;
    __shared__ float red0[128];
    __shared__ float red1[128];
    red0[t] = s; red1[t] = sq;
    __syncthreads();
    for (int off = 64; off > 0; off >>= 1) {
        if (t < off) { red0[t] += red0[t + off]; red1[t] += red1[t + off]; }
        __syncthreads();
    }
    float mean = red0[0] * (1.0f / DIM_);
    float var  = red1[0] * (1.0f / DIM_) - mean * mean;
    float rstd = rsqrtf(var + 1e-6f);
    _Float16* dr = dst + (size_t)row * DIM_;
    float vv[4] = {v4.x, v4.y, v4.z, v4.w};
#pragma unroll
    for (int j = 0; j < 4; j++) {
        int cidx = t * 4 + j;
        float y = (vv[j] - mean) * rstd * scale[cidx] + bias[cidx];
        dr[cidx] = (_Float16)y;
    }
}

__global__ __launch_bounds__(128)
void scatter_residual_kernel(const float* __restrict__ x,
                             const float* __restrict__ projout,
                             float* __restrict__ x2) {
    int row = blockIdx.x;              // canvas token row
    int b = row / L_;
    int l = row % L_;
    int c = l / (H_ * W_);
    int h = (l / W_) % H_;
    int w = l % W_;
    int cs = (c - SS0_ + C_) % C_;
    int hs = (h - SS1_ + H_) % H_;
    int ws = (w - SS2_ + W_) % W_;
    int c1 = cs / WS0, i0 = cs % WS0;
    int h1 = hs / WS1, i1 = hs % WS1;
    int w1 = ws / WS2, i2 = ws % WS2;
    int wt = b * (NW_ * N_TOK) + ((c1 * H1_ + h1) * W1_ + w1) * N_TOK +
             (i0 * WS1 * WS2 + i1 * WS2 + i2);
    const float4* src = (const float4*)(projout + (size_t)wt * DIM_);
    const float4* xr  = (const float4*)(x + (size_t)row * DIM_);
    float4* dr        = (float4*)(x2 + (size_t)row * DIM_);
    int d = threadIdx.x;               // 128 threads, DIM_/4 = 128 float4
    float4 a = xr[d], p = src[d];
    dr[d] = make_float4(a.x + p.x, a.y + p.y, a.z + p.z, a.w + p.w);
}

// -------------------- WMMA GEMM (async double-buffered) --------------------

#define BM 64
#define BN 128
#define BK 32

#define EPI_F16_BIAS      0
#define EPI_F16_BIAS_GELU 1
#define EPI_F32_BIAS      2
#define EPI_F32_BIAS_RES  3

// stage one K-step: A 64x32 (256 x 16B chunks), B 32x128 (512 chunks); 3 async
// ops per thread (1 for A, 2 for B)
__device__ __forceinline__ void stage_tiles(const _Float16* __restrict__ A,
                                            const _Float16* __restrict__ Bw,
                                            _Float16 (*As)[BK],
                                            _Float16 (*Bs)[BN],
                                            int bm, int bn, int k0, int K,
                                            int N, int tid) {
    {
        int r = tid >> 2;               // 4 chunks per 32-half row
        int c = (tid & 3) * 8;
        async_copy16(&As[r][c], A + (size_t)(bm + r) * K + k0 + c);
    }
#pragma unroll
    for (int ph = 0; ph < 2; ph++) {
        int ch = tid + ph * 256;
        int r = ch >> 4;                // 16 chunks per 128-half row
        int c = (ch & 15) * 8;
        async_copy16(&Bs[r][c], Bw + (size_t)(k0 + r) * N + bn + c);
    }
}

template <int EPI>
__global__ __launch_bounds__(256)
void gemm_wmma_kernel(const _Float16* __restrict__ A,
                      const _Float16* __restrict__ Bw,
                      const float* __restrict__ bias,
                      _Float16* __restrict__ outH,
                      float* __restrict__ outF,
                      const float* __restrict__ resid,
                      int M, int N, int K) {
    __shared__ _Float16 As[2][BM][BK];
    __shared__ _Float16 Bs[2][BK][BN];
    int tid  = threadIdx.x;
    int lane = tid & 31;
    int wave = tid >> 5;     // 0..7
    int wm   = wave >> 2;    // 0..1
    int wn   = wave & 3;     // 0..3
    int bm = blockIdx.y * BM;
    int bn = blockIdx.x * BN;
    (void)M;

    v8f acc[2][2];
#pragma unroll
    for (int i = 0; i < 2; i++)
#pragma unroll
        for (int j = 0; j < 2; j++)
            acc[i][j] = (v8f){0.f, 0.f, 0.f, 0.f, 0.f, 0.f, 0.f, 0.f};

    int nsteps = K / BK;
    stage_tiles(A, Bw, As[0], Bs[0], bm, bn, 0, K, N, tid);

    for (int s = 0; s < nsteps; s++) {
        int cur = s & 1;
        if (s + 1 < nsteps) {
            stage_tiles(A, Bw, As[cur ^ 1], Bs[cur ^ 1], bm, bn, (s + 1) * BK,
                        K, N, tid);
            WAIT_ASYNC(3);   // keep next stage's 3 ops in flight
        } else {
            WAIT_ASYNC(0);
        }
        __syncthreads();
        v16h afrag[2], bfrag[2];
#pragma unroll
        for (int i = 0; i < 2; i++)
            afrag[i] = load_a_frag(&As[cur][wm * 32 + i * 16][0], BK, lane);
#pragma unroll
        for (int j = 0; j < 2; j++)
            bfrag[j] = load_b_frag(&Bs[cur][0][wn * 32 + j * 16], BN, lane);
#pragma unroll
        for (int i = 0; i < 2; i++)
#pragma unroll
            for (int j = 0; j < 2; j++)
                acc[i][j] = __builtin_amdgcn_wmma_f32_16x16x32_f16(
                    false, afrag[i], false, bfrag[j], (short)0, acc[i][j],
                    false, false);
        __syncthreads();
    }

    int n  = lane & 15;
    int hi = lane >> 4;
#pragma unroll
    for (int i = 0; i < 2; i++) {
#pragma unroll
        for (int j = 0; j < 2; j++) {
            int col = bn + wn * 32 + j * 16 + n;
            float bsv = bias[col];
#pragma unroll
            for (int r = 0; r < 8; r++) {
                int row = bm + wm * 32 + i * 16 + r + hi * 8;
                float val = acc[i][j][r] + bsv;
                if (EPI == EPI_F16_BIAS) {
                    outH[(size_t)row * N + col] = (_Float16)val;
                } else if (EPI == EPI_F16_BIAS_GELU) {
                    float x3 = val * val * val;
                    float g = 0.5f * val *
                              (1.f + tanhf(0.7978845608f * (val + 0.044715f * x3)));
                    outH[(size_t)row * N + col] = (_Float16)g;
                } else if (EPI == EPI_F32_BIAS) {
                    outF[(size_t)row * N + col] = val;
                } else {
                    outF[(size_t)row * N + col] =
                        val + resid[(size_t)row * N + col];
                }
            }
        }
    }
}

// -------------------- attention --------------------
// one block per (window, head): 9 waves, wave wi owns query rows wi*16..wi*16+15
__global__ __launch_bounds__(288)
void attn_kernel(const _Float16* __restrict__ qkv,   // [ROWS][1536]
                 _Float16* __restrict__ attn_out) {  // [ROWS][512]
    __shared__ _Float16 Ks[N_TOK][DH_];
    __shared__ _Float16 Vs[160][DH_];        // K padded to 160 with zeros
    __shared__ _Float16 Pbuf[9][16][160];    // S (f16) then P=exp(S-max)
    __shared__ float    Inv[9][16];
    __shared__ int      Grp[N_TOK];

    int blk  = blockIdx.x;
    int head = blk % HEADS_;
    int win  = blk / HEADS_;          // 0..B*NW-1
    int winBase = win * N_TOK;
    int wIdx = win % NW_;
    int c1 = wIdx / (H1_ * W1_);
    int h1 = (wIdx / W1_) % H1_;
    int w1 = wIdx % W1_;

    int tid  = threadIdx.x;
    int lane = tid & 31;
    int wi   = tid >> 5;              // 0..8

    // async stage K, V (this head): 1152 16B chunks each, 288 threads -> 4 per
    for (int ch = tid; ch < N_TOK * DH_ / 8; ch += 288) {
        int nr = ch >> 3, d = (ch & 7) * 8;
        size_t ro = (size_t)(winBase + nr) * 1536 + head * DH_ + d;
        async_copy16(&Ks[nr][d], qkv + ro + 512);
        async_copy16(&Vs[nr][d], qkv + ro + 1024);
    }
    for (int i = tid; i < 16 * DH_; i += 288)
        Vs[N_TOK + i / DH_][i % DH_] = (_Float16)0;
    for (int p = tid; p < N_TOK; p += 288) {
        int i0 = p / (WS1 * WS2);
        int i1 = (p / WS2) % WS1;
        int i2 = p % WS2;
        Grp[p] = group_label(c1 * WS0 + i0, h1 * WS1 + i1, w1 * WS2 + i2);
    }
    WAIT_ASYNC(0);
    __syncthreads();

    const float scale = 0.125f;       // 1/sqrt(64)
    int mBase = wi * 16;
    int n   = lane & 15;
    int hi2 = lane >> 4;

    // Q fragments (two K=32 steps covering DH=64): 4x b128 from global
    v16h qa[2];
    {
        int m = lane & 15;
        const _Float16* qrow =
            qkv + (size_t)(winBase + mBase + m) * 1536 + head * DH_ + hi2 * 8;
        union { uint4 q[2]; v16h v; } u0, u1;
        u0.q[0] = *(const uint4*)(qrow);
        u0.q[1] = *(const uint4*)(qrow + 16);
        u1.q[0] = *(const uint4*)(qrow + 32);
        u1.q[1] = *(const uint4*)(qrow + 48);
        qa[0] = u0.v;
        qa[1] = u1.v;
    }

    // S = Q K^T * scale + mask  (9 column tiles of 16)
    for (int j = 0; j < 9; j++) {
        v8f acc = (v8f){0.f, 0.f, 0.f, 0.f, 0.f, 0.f, 0.f, 0.f};
#pragma unroll
        for (int s2 = 0; s2 < 2; s2++) {
            // B[kk][nn] = Ks[j*16+nn][kk]: two 8-half runs per lane (k = 8h.., 16+8h..)
            const _Float16* krow = &Ks[j * 16 + n][s2 * 32 + hi2 * 8];
            union { uint4 q[2]; v16h v; } ub;
            ub.q[0] = *(const uint4*)(krow);
            ub.q[1] = *(const uint4*)(krow + 16);
            acc = __builtin_amdgcn_wmma_f32_16x16x32_f16(
                false, qa[s2], false, ub.v, (short)0, acc, false, false);
        }
#pragma unroll
        for (int r = 0; r < 8; r++) {
            int lm = r + hi2 * 8;
            int kj = j * 16 + n;
            float sv = acc[r] * scale;
            if (Grp[mBase + lm] != Grp[kj]) sv = -60000.0f;  // -inf mask (f16 safe)
            Pbuf[wi][lm][kj] = (_Float16)sv;
        }
    }
    __syncthreads();

    // softmax over 144 cols per row: 2 lanes per row, 72 cols each
    {
        int rrow = lane >> 1;
        int half = lane & 1;
        _Float16* prow = &Pbuf[wi][rrow][0];
        float mx = -3.0e38f;
        for (int c2 = half * 72; c2 < half * 72 + 72; c2++)
            mx = fmaxf(mx, (float)prow[c2]);
        mx = fmaxf(mx, __shfl_xor(mx, 1, 32));
        float sum = 0.f;
        for (int c2 = half * 72; c2 < half * 72 + 72; c2++) {
            float e = __expf((float)prow[c2] - mx);
            sum += e;
            prow[c2] = (_Float16)e;
        }
        sum += __shfl_xor(sum, 1, 32);
        if (half == 0) {
            Inv[wi][rrow] = 1.f / sum;
            for (int c2 = N_TOK; c2 < 160; c2++) prow[c2] = (_Float16)0;
        }
    }
    __syncthreads();

    // O = P V (K=160 over 5 steps, 4 output column tiles), scale rows by 1/sum
    for (int nj = 0; nj < 4; nj++) {
        v8f acc = (v8f){0.f, 0.f, 0.f, 0.f, 0.f, 0.f, 0.f, 0.f};
#pragma unroll
        for (int ks = 0; ks < 5; ks++) {
            v16h af = load_a_frag(&Pbuf[wi][0][ks * 32], 160, lane);
            v16h bf = load_b_frag(&Vs[ks * 32][nj * 16], DH_, lane);
            acc = __builtin_amdgcn_wmma_f32_16x16x32_f16(
                false, af, false, bf, (short)0, acc, false, false);
        }
#pragma unroll
        for (int r = 0; r < 8; r++) {
            int lm = r + hi2 * 8;
            float o = acc[r] * Inv[wi][lm];
            attn_out[(size_t)(winBase + mBase + lm) * DIM_ + head * DH_ +
                     nj * 16 + n] = (_Float16)o;
        }
    }
}

// -------------------- launch --------------------

extern "C" void kernel_launch(void* const* d_in, const int* in_sizes, int n_in,
                              void* d_out, int out_size, void* d_ws,
                              size_t ws_size, hipStream_t stream) {
    const float* x      = (const float*)d_in[0];
    const float* qkv_w  = (const float*)d_in[1];
    const float* qkv_b  = (const float*)d_in[2];
    const float* proj_w = (const float*)d_in[3];
    const float* proj_b = (const float*)d_in[4];
    const float* n1s    = (const float*)d_in[5];
    const float* n1b    = (const float*)d_in[6];
    const float* n2s    = (const float*)d_in[7];
    const float* n2b    = (const float*)d_in[8];
    const float* mlp_w1 = (const float*)d_in[9];
    const float* mlp_b1 = (const float*)d_in[10];
    const float* mlp_w2 = (const float*)d_in[11];
    const float* mlp_b2 = (const float*)d_in[12];
    float* out = (float*)d_out;

    char* ws = (char*)d_ws;
    size_t off = 0;
    auto alloc = [&](size_t bytes) -> void* {
        off = (off + 255) & ~(size_t)255;
        void* p = ws + off;
        off += bytes;
        return p;
    };

    _Float16* wqkv_h = (_Float16*)alloc((size_t)512 * 1536 * 2);
    _Float16* wproj_h = (_Float16*)alloc((size_t)512 * 512 * 2);
    _Float16* wm1_h = (_Float16*)alloc((size_t)512 * 2048 * 2);
    _Float16* wm2_h = (_Float16*)alloc((size_t)2048 * 512 * 2);
    _Float16* hwin = (_Float16*)alloc((size_t)ROWS * DIM_ * 2);   // LN1 / LN2 out
    _Float16* big  = (_Float16*)alloc((size_t)ROWS * 2048 * 2);   // qkv+attn / hidden
    _Float16* qkvb = big;                                         // [ROWS][1536]
    _Float16* attnout = big + (size_t)ROWS * 1536;                // [ROWS][512]
    _Float16* hidden = big;                                       // [ROWS][2048]
    float* projout = (float*)alloc((size_t)ROWS * DIM_ * 4);
    float* x2      = (float*)alloc((size_t)ROWS * DIM_ * 4);
    (void)ws_size; (void)in_sizes; (void)n_in; (void)out_size;

    // 1. weights fp32 -> f16
    f32_to_f16_kernel<<<(512 * 1536 + 255) / 256, 256, 0, stream>>>(qkv_w, wqkv_h, 512 * 1536);
    f32_to_f16_kernel<<<(512 * 512 + 255) / 256, 256, 0, stream>>>(proj_w, wproj_h, 512 * 512);
    f32_to_f16_kernel<<<(512 * 2048 + 255) / 256, 256, 0, stream>>>(mlp_w1, wm1_h, 512 * 2048);
    f32_to_f16_kernel<<<(2048 * 512 + 255) / 256, 256, 0, stream>>>(mlp_w2, wm2_h, 2048 * 512);

    // 2. LN1 + shift + window partition
    ln_gather_kernel<<<ROWS, 128, 0, stream>>>(x, n1s, n1b, hwin, 0);

    // 3. QKV GEMM
    gemm_wmma_kernel<EPI_F16_BIAS><<<dim3(1536 / BN, ROWS / BM), 256, 0, stream>>>(
        hwin, wqkv_h, qkv_b, qkvb, nullptr, nullptr, ROWS, 1536, 512);

    // 4. windowed multi-head attention
    attn_kernel<<<B_ * NW_ * HEADS_, 288, 0, stream>>>(qkvb, attnout);

    // 5. proj GEMM (f32 out)
    gemm_wmma_kernel<EPI_F32_BIAS><<<dim3(512 / BN, ROWS / BM), 256, 0, stream>>>(
        attnout, wproj_h, proj_b, nullptr, projout, nullptr, ROWS, 512, 512);

    // 6. window reverse + un-shift + residual
    scatter_residual_kernel<<<ROWS, 128, 0, stream>>>(x, projout, x2);

    // 7. LN2
    ln_gather_kernel<<<ROWS, 128, 0, stream>>>(x2, n2s, n2b, hwin, 1);

    // 8. MLP fc1 + GELU
    gemm_wmma_kernel<EPI_F16_BIAS_GELU><<<dim3(2048 / BN, ROWS / BM), 256, 0, stream>>>(
        hwin, wm1_h, mlp_b1, hidden, nullptr, nullptr, ROWS, 2048, 512);

    // 9. MLP fc2 + residual -> d_out
    gemm_wmma_kernel<EPI_F32_BIAS_RES><<<dim3(512 / BN, ROWS / BM), 256, 0, stream>>>(
        hidden, wm2_h, mlp_b2, nullptr, out, x2, ROWS, 512, 2048);
}
